// SubLightGru_39728447488308
// MI455X (gfx1250) — matrix operand order repaired and checked by reading, not compiled, not run on hardware
//
#include <hip/hip_runtime.h>
#include <hip/hip_bf16.h>

typedef __attribute__((ext_vector_type(16))) _Float16 v16h;
typedef __attribute__((ext_vector_type(8)))  _Float16 v8h;
typedef __attribute__((ext_vector_type(8)))  float    v8f;

union V16u { v16h v; v8h h[2]; _Float16 e[16]; };

#define B_    8
#define C_    64
#define T_    300
#define D_    161
#define H_    128
#define O_    64
#define N_    (B_ * D_)        // 1288
#define TD_   (T_ * D_)        // 48300
#define TWOH_ 256
#define EPS_  1e-5f

// ws layout (f16 weights pre-swizzled into WMMA B-operand order)
#define WX_TILES 32            // 2 K-chunks x 16 col-tiles
#define WH_TILES 64            // 4 K-chunks x 16 col-tiles
#define WO_TILES 16            // 4 K-chunks x 4 col-tiles
#define WX_OFF   0
#define WH_OFF   (WX_TILES * 512)
#define WO_OFF   ((WX_TILES + WH_TILES) * 512)
#define WTOT     ((WX_TILES + WH_TILES + WO_TILES) * 512)   // 57344 halves

// BN scratch (float offsets into ws, starting at byte 131072)
#define TC_       16                      // time chunks for partial stats
#define TCHUNK_   19                      // ceil(300/16)
#define PSUM_OFF  (131072 / 4)
#define PSZ_      (TC_ * O_ * D_)         // 164864 floats
#define PSSQ_OFF  (PSUM_OFF + PSZ_)
#define SUM_OFF   (PSSQ_OFF + PSZ_)
#define SSQ_OFF   (SUM_OFF + O_ * D_)
#define CNT_OFF   (SSQ_OFF + O_ * D_)

// LDS row strides padded to dodge bank conflicts
#define HSTR 136               // halves per h row (128 + 8), 272B, 16B-aligned
#define YSTR 17                // floats per y row in wave-private staging

__device__ __forceinline__ v8f wmma16(v16h a, v16h b, v8f c) {
    return __builtin_amdgcn_wmma_f32_16x16x32_f16(false, a, false, b, (short)0, c, false, false);
}

__device__ __forceinline__ v16h ldb(const _Float16* __restrict__ w, int tileIdx, int lane) {
    const v8h* p = (const v8h*)(w + tileIdx * 512 + lane * 16);
    V16u r; r.h[0] = p[0]; r.h[1] = p[1];
    return r.v;
}

__device__ __forceinline__ float fast_sigmoid(float x) {
    // 1/(1+exp(-x)) with v_exp2 + v_rcp (branch-free)
    return __builtin_amdgcn_rcpf(1.f + __builtin_amdgcn_exp2f(x * -1.44269504088896340736f));
}

__device__ __forceinline__ float fast_tanh(float x) {
#if __has_builtin(__builtin_amdgcn_tanhf)
    return __builtin_amdgcn_tanhf(x);
#else
    float r;
    asm("v_tanh_f32 %0, %1" : "=v"(r) : "v"(x));
    return r;
#endif
}

// ---------------- weight prep: f32 -> f16, swizzled to B-operand layout ----------------
__global__ void prep_weights(const float* __restrict__ Wx, const float* __restrict__ Wh,
                             const float* __restrict__ Wo, _Float16* __restrict__ wsw) {
    int idx = blockIdx.x * 256 + threadIdx.x;
    if (idx >= WTOT) return;
    if (idx < WH_OFF) {                       // Wx [64 x 256]
        int tile = idx >> 9, r = idx & 511;
        int kc = tile >> 4, nt = tile & 15;
        int lane = r >> 4, j = r & 15;
        int k = kc * 32 + (j >> 3) * 16 + (lane >> 4) * 8 + (j & 7);
        int n = nt * 16 + (lane & 15);
        wsw[idx] = (_Float16)Wx[k * TWOH_ + n];
    } else if (idx < WO_OFF) {                // Wh [128 x 256]
        int l = idx - WH_OFF;
        int tile = l >> 9, r = l & 511;
        int kc = tile >> 4, nt = tile & 15;
        int lane = r >> 4, j = r & 15;
        int k = kc * 32 + (j >> 3) * 16 + (lane >> 4) * 8 + (j & 7);
        int n = nt * 16 + (lane & 15);
        wsw[idx] = (_Float16)Wh[k * TWOH_ + n];
    } else {                                  // Wo [128 x 64]
        int l = idx - WO_OFF;
        int tile = l >> 9, r = l & 511;
        int kc = tile >> 2, ot = tile & 3;
        int lane = r >> 4, j = r & 15;
        int k = kc * 32 + (j >> 3) * 16 + (lane >> 4) * 8 + (j & 7);
        int n = ot * 16 + (lane & 15);
        wsw[idx] = (_Float16)Wo[k * O_ + n];
    }
}

// ---------------- fused GRU: x@Wx + h@Wh + bias -> gates -> h -> h@Wo -> y ----------------
// 256 threads = 8 waves; block owns 16 rows of N; wave w owns z/c column-pair p=w.
// Waves 0..3 additionally produce the 4 O-tiles of y, pipelined one step behind so the
// y GEMM overlaps the recurrence WMMAs instead of extending the barrier critical path.
__global__ __launch_bounds__(256) void gru_kernel(
    const float* __restrict__ inp, const int* __restrict__ lens,
    const float* __restrict__ bvec, const float* __restrict__ bo,
    const _Float16* __restrict__ wsw, float* __restrict__ out)
{
    __shared__ _Float16 hbuf[2][16 * HSTR];       // double-buffered h (f16)
    __shared__ float    ybuf[4][16 * YSTR];       // wave-private y staging
    __shared__ int sRowIn[16], sRowOut[16], sLen[16];

    const int tid  = threadIdx.x;
    const int lane = tid & 31;
    const int wid  = tid >> 5;      // 0..7
    const int col  = lane & 15;
    const int hi   = lane >> 4;
    const int nbase = blockIdx.x * 16;

    if (tid < 16) {
        int n  = nbase + tid;
        int ok = (n < N_);
        int nc = ok ? n : (N_ - 1);
        int bb = nc / D_;
        int dd = nc - bb * D_;
        sRowIn[tid]  = bb * C_ * TD_ + dd;
        sRowOut[tid] = bb * O_ * TD_ + dd;
        sLen[tid]    = ok ? lens[bb] : 0;
    }
    __syncthreads();

    const int rib = sRowIn[col];
    int lenv[8];
#pragma unroll
    for (int i = 0; i < 8; ++i) lenv[i] = sLen[hi * 8 + i];
    const bool rowStoreOK = (nbase + (tid & 15)) < N_;
    const int  robm = sRowOut[tid & 15];

    const int p = wid;                                   // this wave's column pair
    const float bz  = bvec[p * 16 + col];
    const float bc  = bvec[128 + p * 16 + col];
    const float bov = bo[(wid & 3) * 16 + col];

    const _Float16* wx = wsw + WX_OFF;
    const _Float16* wh = wsw + WH_OFF;
    const _Float16* wo = wsw + WO_OFF;

    // x gather bases: 4 groups of 8 consecutive-K loads -> immediate offsets j*TD*4
    const float* xb0 = inp + rib + (hi * 8) * TD_;
    const float* xb1 = inp + rib + (16 + hi * 8) * TD_;
    const float* xb2 = inp + rib + (32 + hi * 8) * TD_;
    const float* xb3 = inp + rib + (48 + hi * 8) * TD_;

    v16h hA[4]; hA[0] = (v16h){}; hA[1] = (v16h){}; hA[2] = (v16h){}; hA[3] = (v16h){};
    v8f  hD = {};

    // y(t) = h(t) @ Wo + bo for waves 0..3; Wo tiles deliberately re-loaded (laundered
    // pointer) so they stay out of the persistent VGPR budget.
    auto do_y = [&](int ty) {
        const _Float16* wop = wo;
        asm volatile("" : "+s"(wop));
        v8f ya;
#pragma unroll
        for (int i = 0; i < 8; ++i) ya[i] = bov;
#pragma unroll
        for (int kc = 0; kc < 4; ++kc)
            ya = wmma16(hA[kc], ldb(wop, kc * 4 + wid, lane), ya);
        float* yb = ybuf[wid];
#pragma unroll
        for (int i = 0; i < 8; ++i)
            yb[(i + 8 * hi) * YSTR + col] = (ty < lenv[i]) ? ya[i] : 0.f;
        // in-wave DS ordering: no barrier needed before reading own staging
#pragma unroll
        for (int it = 0; it < 8; ++it) {
            int oo = it * 2 + hi;
            if (rowStoreOK)
                out[robm + (wid * 16 + oo) * TD_ + ty * D_] = yb[col * YSTR + oo];
        }
    };

    // prologue: x tile for t = 0
    V16u xa0, xa1;
#pragma unroll
    for (int j = 0; j < 8; ++j) {
        xa0.e[j]     = (_Float16)xb0[j * TD_];
        xa0.e[8 + j] = (_Float16)xb1[j * TD_];
        xa1.e[j]     = (_Float16)xb2[j * TD_];
        xa1.e[8 + j] = (_Float16)xb3[j * TD_];
    }

#pragma unroll 1
    for (int t = 0; t < T_; ++t) {
        // ---- issue next step's x loads early (latency hidden behind WMMAs) ----
        V16u nx0, nx1;
        if (t + 1 < T_) {
            const int nD = (t + 1) * D_;
#pragma unroll
            for (int j = 0; j < 8; ++j) {
                nx0.e[j]     = (_Float16)xb0[nD + j * TD_];
                nx0.e[8 + j] = (_Float16)xb1[nD + j * TD_];
                nx1.e[j]     = (_Float16)xb2[nD + j * TD_];
                nx1.e[8 + j] = (_Float16)xb3[nD + j * TD_];
            }
            __builtin_prefetch(xb0 + nD + D_, 0, 3);
        }

        // ---- gates: acc = bias + x@Wx + h@Wh  (hA holds h(t-1)) ----
        v8f az, ac;
#pragma unroll
        for (int i = 0; i < 8; ++i) { az[i] = bz; ac[i] = bc; }
        az = wmma16(xa0.v, ldb(wx, 0 * 16 + p,     lane), az);
        az = wmma16(xa1.v, ldb(wx, 1 * 16 + p,     lane), az);
        ac = wmma16(xa0.v, ldb(wx, 0 * 16 + p + 8, lane), ac);
        ac = wmma16(xa1.v, ldb(wx, 1 * 16 + p + 8, lane), ac);
#pragma unroll
        for (int kc = 0; kc < 4; ++kc) {
            az = wmma16(hA[kc], ldb(wh, kc * 16 + p,     lane), az);
            ac = wmma16(hA[kc], ldb(wh, kc * 16 + p + 8, lane), ac);
        }

        // ---- y(t-1): independent of the accumulation chain; overlaps recurrence ----
        if (wid < 4 && t > 0) do_y(t - 1);

        // ---- state update (branch-free activations), publish h(t) as f16 ----
        _Float16* hb = hbuf[t & 1];
#pragma unroll
        for (int i = 0; i < 8; ++i) {
            float z  = fast_sigmoid(az[i]);
            float hc = fast_tanh(ac[i]);
            float hn = z * hD[i] + (1.f - z) * hc;
            hn = (t < lenv[i]) ? hn : hD[i];     // freeze past sequence end
            hD[i] = hn;
            hb[(i + 8 * hi) * HSTR + p * 16 + col] = (_Float16)hn;
        }

        __syncthreads();   // single barrier per step (hbuf double-buffered)

        // ---- D-layout -> A-layout transpose of h(t) via LDS ----
#pragma unroll
        for (int kc = 0; kc < 4; ++kc) {
            V16u a;
            a.h[0] = *(const v8h*)&hb[col * HSTR + kc * 32 + hi * 8];
            a.h[1] = *(const v8h*)&hb[col * HSTR + kc * 32 + 16 + hi * 8];
            hA[kc] = a.v;
        }

        xa0 = nx0; xa1 = nx1;
    }

    // epilogue: y(T-1) from the final hidden state
    if (wid < 4) do_y(T_ - 1);
}

// ---------------- masked BN: two-stage stats + in-place apply ----------------
__global__ void bn_partial(const float* __restrict__ out, const int* __restrict__ lens,
                           float* __restrict__ psum, float* __restrict__ pssq) {
    int o = blockIdx.x, tc = blockIdx.y, d = threadIdx.x;
    if (d >= D_) return;
    int tlo = tc * TCHUNK_;
    int thi = min(tlo + TCHUNK_, T_);
    float s = 0.f, q = 0.f;
    for (int b = 0; b < B_; ++b) {
        int te = min(thi, lens[b]);
        const float* p = out + (size_t)(b * O_ + o) * TD_ + d;
        for (int t = tlo; t < te; ++t) { float v = p[t * D_]; s += v; q += v * v; }
    }
    psum[(tc * O_ + o) * D_ + d] = s;
    pssq[(tc * O_ + o) * D_ + d] = q;
}

__global__ void bn_reduce(const float* __restrict__ psum, const float* __restrict__ pssq,
                          const int* __restrict__ lens,
                          float* __restrict__ sums, float* __restrict__ ssqs,
                          float* __restrict__ cnt) {
    int o = blockIdx.x, d = threadIdx.x;
    if (d >= D_) return;
    float s = 0.f, q = 0.f;
#pragma unroll
    for (int tc = 0; tc < TC_; ++tc) {
        s += psum[(tc * O_ + o) * D_ + d];
        q += pssq[(tc * O_ + o) * D_ + d];
    }
    sums[o * D_ + d] = s;
    ssqs[o * D_ + d] = q;
    if (o == 0 && d == 0) {
        float c = 0.f;
#pragma unroll
        for (int b = 0; b < B_; ++b) c += (float)lens[b];
        *cnt = c;
    }
}

__global__ void bn_apply(float* __restrict__ out, const int* __restrict__ lens,
                         const float* __restrict__ gamma, const float* __restrict__ beta,
                         const float* __restrict__ sums, const float* __restrict__ ssqs,
                         const float* __restrict__ cnt) {
    long idx = (long)blockIdx.x * blockDim.x + threadIdx.x;
    const long total = (long)B_ * O_ * T_ * D_;
    if (idx >= total) return;
    int d = (int)(idx % D_);
    long r = idx / D_;
    int t = (int)(r % T_);
    long r2 = r / T_;
    int o = (int)(r2 % O_);
    int b = (int)(r2 / O_);
    if (t >= lens[b]) { out[idx] = 0.f; return; }
    float c = *cnt;
    int ch = o * D_ + d;
    float mean = sums[ch] / c;
    float var  = ssqs[ch] / c - mean * mean;
    float v = out[idx];
    out[idx] = (v - mean) * rsqrtf(var + EPS_) * gamma[ch] + beta[ch];
}

extern "C" void kernel_launch(void* const* d_in, const int* in_sizes, int n_in,
                              void* d_out, int out_size, void* d_ws, size_t ws_size,
                              hipStream_t stream) {
    const float* inp   = (const float*)d_in[0];
    const int*   lens  = (const int*)  d_in[1];
    const float* Wx    = (const float*)d_in[2];
    const float* Wh    = (const float*)d_in[3];
    const float* bvec  = (const float*)d_in[4];
    const float* Wo    = (const float*)d_in[5];
    const float* bo    = (const float*)d_in[6];
    const float* gamma = (const float*)d_in[7];
    const float* beta  = (const float*)d_in[8];
    float*       out   = (float*)d_out;
    _Float16*    wsw   = (_Float16*)d_ws;
    float*       wsf   = (float*)d_ws;

    prep_weights<<<(WTOT + 255) / 256, 256, 0, stream>>>(Wx, Wh, Wo, wsw);
    gru_kernel<<<(N_ + 15) / 16, 256, 0, stream>>>(inp, lens, bvec, bo, wsw, out);
    bn_partial<<<dim3(O_, TC_), 192, 0, stream>>>(out, lens, wsf + PSUM_OFF, wsf + PSSQ_OFF);
    bn_reduce<<<O_, 192, 0, stream>>>(wsf + PSUM_OFF, wsf + PSSQ_OFF, lens,
                                      wsf + SUM_OFF, wsf + SSQ_OFF, wsf + CNT_OFF);
    const long total = (long)B_ * O_ * T_ * D_;
    bn_apply<<<(int)((total + 255) / 256), 256, 0, stream>>>(out, lens, gamma, beta,
                                                             wsf + SUM_OFF, wsf + SSQ_OFF,
                                                             wsf + CNT_OFF);
}
